// EqvRESFeedForward_51994874085883
// MI455X (gfx1250) — matrix operand order repaired and checked by reading, not compiled, not run on hardware
//
#include <hip/hip_runtime.h>
#include <hip/hip_bf16.h>

typedef __attribute__((ext_vector_type(16))) _Float16 v16h;
typedef __attribute__((ext_vector_type(8)))  _Float16 v8h;
typedef __attribute__((ext_vector_type(8)))  float    v8f;

#define BN   2
#define NN   384
#define CC   16
#define NB_  10
#define HH   64
#define KTOT (NN * HH)   // 24576 (K index k = m*64 + h)

__device__ __forceinline__ float swish_f(float v) {
    return v / (1.0f + __expf(-v));
}

// ---------------------------------------------------------------------------
// hid[b,n,m,h] = swish( sum_k rbf_k(r_nm) * w1[k,h] ), stored f16, h contiguous
// ---------------------------------------------------------------------------
__global__ void __launch_bounds__(256)
hid_kernel(const float* __restrict__ xyz, const float* __restrict__ w1,
           _Float16* __restrict__ hid)
{
    __shared__ float sw1[NB_ * HH];
    for (int t = threadIdx.x; t < NB_ * HH; t += blockDim.x) sw1[t] = w1[t];
    __syncthreads();

    int idx = blockIdx.x * blockDim.x + threadIdx.x;   // total = BN*NN*NN (exact)
    int m = idx % NN; int t = idx / NN;
    int n = t % NN;   int b = t / NN;

    const float* pn = xyz + (size_t)(b * NN + n) * 3;
    const float* pm = xyz + (size_t)(b * NN + m) * 3;
    float dx = pn[0] - pm[0], dy = pn[1] - pm[1], dz = pn[2] - pm[2];
    float r = sqrtf(dx * dx + dy * dy + dz * dz + 1e-12f);

    float rbf[NB_];
    const float step = 10.0f / 9.0f;   // linspace(0,10,10) spacing; WIDTH = 1.0
    #pragma unroll
    for (int k = 0; k < NB_; k++) { float u = r - step * (float)k; rbf[k] = __expf(-u * u); }

    _Float16* outp = hid + (size_t)idx * HH;
    #pragma unroll 8
    for (int h = 0; h < HH; h++) {
        float a = 0.f;
        #pragma unroll
        for (int k = 0; k < NB_; k++) a += rbf[k] * sw1[k * HH + h];
        outp[h] = (_Float16)swish_f(a);
    }
}

// ---------------------------------------------------------------------------
// Gt[b, i, m*64+h] = sum_j w2[h, i*16+j] * xin[b,m,j]   (f16, K-major for WMMA B)
// use_mask: zero rows where diag(mask) == 0 (conv1 input masking)
// ---------------------------------------------------------------------------
__global__ void __launch_bounds__(256)
g_kernel(const float* __restrict__ xin, const int* __restrict__ mask,
         const float* __restrict__ w2, _Float16* __restrict__ Gt, int use_mask)
{
    int idx = blockIdx.x * blockDim.x + threadIdx.x;   // total = BN*NN*HH*CC (exact)
    int i = idx & 15;  int t = idx >> 4;
    int h = t & 63;    t >>= 6;
    int m = t % NN;    int b = t / NN;

    float keep = 1.0f;
    if (use_mask)
        keep = (mask[((size_t)(b * NN + m)) * NN + m] != 0) ? 1.0f : 0.0f;

    const float* xr = xin + (size_t)(b * NN + m) * CC;
    const float* wr = w2 + (size_t)h * (CC * CC) + i * CC;
    float acc = 0.f;
    #pragma unroll
    for (int j = 0; j < CC; j++) acc += wr[j] * xr[j];

    Gt[((size_t)(b * CC + i)) * KTOT + (size_t)m * HH + h] = (_Float16)(acc * keep);
}

// ---------------------------------------------------------------------------
// xout[b, n, i] = (1/sqrt(N)) * sum_k hid[b,n,k] * Gt[b,i,k]
// One workgroup per (b, 16-row n-tile). 8 waves split K, LDS reduce.
// v_wmma_f32_16x16x32_f16, f32 accumulation.
// ---------------------------------------------------------------------------
__global__ void __launch_bounds__(256)
gemm_kernel(const _Float16* __restrict__ hid, const _Float16* __restrict__ Gt,
            float* __restrict__ xout)
{
    const int n0   = blockIdx.x * 16;
    const int b    = blockIdx.y;
    const int lane = threadIdx.x & 31;
    const int wave = threadIdx.x >> 5;

    // A fragment: lane l, element e -> (M = l&15, K = (e>>3)*16 + (l>>4)*8 + (e&7))
    const _Float16* Abase =
        hid + ((size_t)(b * NN) + n0 + (lane & 15)) * (size_t)KTOT + ((lane >> 4) * 8);
    // B fragment: lane l, element e -> (K = (e>>3)*16 + (l>>4)*8 + (e&7), N = l&15)
    const _Float16* Bbase =
        Gt + ((size_t)(b * CC) + (lane & 15)) * (size_t)KTOT + ((lane >> 4) * 8);

    v8f acc = {};
    const int steps = (KTOT / 32) / 8;   // 96 k-steps per wave
    for (int s = 0; s < steps; s++) {
        const int K0 = (wave + s * 8) * 32;
        const _Float16* ap = Abase + K0;
        const _Float16* bp = Bbase + K0;

        v8h alo = *reinterpret_cast<const v8h*>(ap);        // K..K+7   (this half)
        v8h ahi = *reinterpret_cast<const v8h*>(ap + 16);   // K+16..K+23
        v8h blo = *reinterpret_cast<const v8h*>(bp);
        v8h bhi = *reinterpret_cast<const v8h*>(bp + 16);

        v16h a  = __builtin_shufflevector(alo, ahi, 0,1,2,3,4,5,6,7,8,9,10,11,12,13,14,15);
        v16h bm = __builtin_shufflevector(blo, bhi, 0,1,2,3,4,5,6,7,8,9,10,11,12,13,14,15);

        acc = __builtin_amdgcn_wmma_f32_16x16x32_f16(
            /*neg_a=*/false, a, /*neg_b=*/false, bm,
            /*c_mod=*/(short)0, acc, /*reuse_a=*/false, /*reuse_b=*/false);

        __builtin_prefetch(ap + 8 * 32 * 4, 0, 1);   // 4 of this wave's steps ahead
        __builtin_prefetch(bp + 8 * 32 * 4, 0, 1);
    }

    __shared__ float red[8][256];
    #pragma unroll
    for (int v = 0; v < 8; v++) red[wave][v * 32 + lane] = acc[v];
    __syncthreads();

    const int t = threadIdx.x;
    float s = 0.f;
    #pragma unroll
    for (int w = 0; w < 8; w++) s += red[w][t];

    const int l = t & 31, v = t >> 5;
    const int M = v + ((l >> 4) << 3);   // D layout: VGPR v -> M=v (lanes 0-15), M=v+8 (16-31)
    const int i = l & 15;
    xout[((size_t)(b * NN) + n0 + M) * CC + i] = s * 0.05103103630798288f; // 1/sqrt(384)
}

// ---------------------------------------------------------------------------
// mask-diag, sum over nodes, unbiased layernorm, fc2, softmax. One block per b.
// ---------------------------------------------------------------------------
__global__ void __launch_bounds__(256)
epilogue_kernel(const float* __restrict__ x2, const int* __restrict__ mask,
                const float* __restrict__ fc2, float* __restrict__ out)
{
    const int b = blockIdx.x;
    __shared__ float part[16][16];
    __shared__ float sv[16], yv[16], zv[16];
    __shared__ float smean, sstd;

    const int t = threadIdx.x;
    const int i = t & 15, c = t >> 4;

    float s = 0.f;
    for (int n = c; n < NN; n += 16) {
        if (mask[((size_t)b * NN + n) * NN + n] != 0)
            s += x2[((size_t)b * NN + n) * CC + i];
    }
    part[c][i] = s;
    __syncthreads();

    if (t < 16) {
        float tot = 0.f;
        for (int k = 0; k < 16; k++) tot += part[k][t];
        sv[t] = tot;
    }
    __syncthreads();

    if (t == 0) {
        float mean = 0.f;
        for (int j = 0; j < 16; j++) mean += sv[j];
        mean *= (1.0f / 16.0f);
        float var = 0.f;
        for (int j = 0; j < 16; j++) { float d = sv[j] - mean; var += d * d; }
        var *= (1.0f / 15.0f);          // ddof=1
        smean = mean; sstd = sqrtf(var);
    }
    __syncthreads();

    if (t < 16) yv[t] = (sv[t] - smean) / (sstd + 1e-6f);
    __syncthreads();

    if (t < 16) {
        float z = 0.f;
        for (int j = 0; j < 16; j++) z += yv[j] * fc2[t * 16 + j];  // x @ fc2^T
        zv[t] = z;
    }
    __syncthreads();

    if (t == 0) {
        float mx = zv[0];
        for (int j = 1; j < 16; j++) mx = fmaxf(mx, zv[j]);
        float e[16]; float den = 0.f;
        for (int j = 0; j < 16; j++) { e[j] = __expf(zv[j] - mx); den += e[j]; }
        for (int j = 0; j < 16; j++) out[b * 16 + j] = e[j] / den;
    }
}

// ---------------------------------------------------------------------------
extern "C" void kernel_launch(void* const* d_in, const int* in_sizes, int n_in,
                              void* d_out, int out_size, void* d_ws, size_t ws_size,
                              hipStream_t stream)
{
    (void)in_sizes; (void)n_in; (void)out_size; (void)ws_size;

    const float* x    = (const float*)d_in[0];
    const float* xyz  = (const float*)d_in[1];
    const int*   mask = (const int*)d_in[2];
    const float* w11  = (const float*)d_in[3];
    const float* w12  = (const float*)d_in[4];
    const float* w21  = (const float*)d_in[5];
    const float* w22  = (const float*)d_in[6];
    const float* fc2  = (const float*)d_in[7];
    float* out = (float*)d_out;

    char* ws = (char*)d_ws;
    _Float16* hid = (_Float16*)ws;                                   // 37.75 MB
    size_t off = (size_t)BN * NN * NN * HH * sizeof(_Float16);
    off = (off + 255) & ~(size_t)255;
    _Float16* Gt = (_Float16*)(ws + off);                            // 1.57 MB
    off += (size_t)BN * CC * KTOT * sizeof(_Float16);
    off = (off + 255) & ~(size_t)255;
    float* x1 = (float*)(ws + off);                                  // 48 KB

    dim3 blk(256);
    const int pairBlocks = (BN * NN * NN) / 256;        // 1152
    const int gBlocks    = (BN * NN * HH * CC) / 256;   // 3072
    dim3 ggrid(NN / 16, BN);                            // (24, 2)

    // ---- conv1 ----
    hid_kernel <<<pairBlocks, blk, 0, stream>>>(xyz, w11, hid);
    g_kernel   <<<gBlocks,    blk, 0, stream>>>(x,  mask, w12, Gt, /*use_mask=*/1);
    gemm_kernel<<<ggrid,      blk, 0, stream>>>(hid, Gt, x1);
    // ---- conv2 ----
    hid_kernel <<<pairBlocks, blk, 0, stream>>>(xyz, w21, hid);
    g_kernel   <<<gBlocks,    blk, 0, stream>>>(x1, mask, w22, Gt, /*use_mask=*/0);
    gemm_kernel<<<ggrid,      blk, 0, stream>>>(hid, Gt, x1);        // x1 := conv2 out
    // ---- epilogue ----
    epilogue_kernel<<<BN, blk, 0, stream>>>(x1, mask, fc2, out);
}
